// _FlashRNNCudaLayer_67860483277718
// MI455X (gfx1250) — matrix operand order repaired and checked by reading, not compiled, not run on hardware
//
#include <hip/hip_runtime.h>
#include <hip/hip_bf16.h>

// FlashRNN LSTM forward on gfx1250 (MI455X).
// Grid: NH=12 workgroups (one head each), 128 threads = 4 wave32.
// Wave w owns d-slice [16w,16w+16) across all 4 gates -> register-local cell update.
// Recurrent weights live in VGPRs as bf16 WMMA B-fragments for the whole T loop.
// h is double-buffered in LDS (bf16), one barrier per timestep.
// Wx is software-pipelined one full timestep ahead in registers (NT loads),
// outputs stored non-temporally (both streams ~= L2 capacity).

typedef __bf16 bf16_t;
typedef __attribute__((ext_vector_type(16))) __bf16 v16bf;
typedef __attribute__((ext_vector_type(8)))  __bf16 v8bf;
typedef __attribute__((ext_vector_type(8)))  float  v8f;

#define T_B   16
#define T_NH  12
#define T_NG  4
#define T_D   64
#define HSTR  80   // LDS row stride in elements: 160 bytes, keeps b128 loads 16B-aligned

__device__ __forceinline__ float fsig(float x) {
  return 1.0f / (1.0f + __expf(-x));
}
__device__ __forceinline__ float ftanh_(float x) {
  x = fminf(15.0f, fmaxf(-15.0f, x));
  float e = __expf(2.0f * x);
  return (e - 1.0f) / (e + 1.0f);
}

__launch_bounds__(128, 1)
__global__ void flashrnn_lstm_wmma(const float* __restrict__ Wx,
                                   const float* __restrict__ states,
                                   const float* __restrict__ R,
                                   const float* __restrict__ bias,
                                   float* __restrict__ out,
                                   int T) {
  __shared__ bf16_t hbuf[2 * 16 * HSTR];   // double-buffered h, bf16

  const int head = blockIdx.x;
  const int wave = threadIdx.x >> 5;       // 0..3 : d-slice owner
  const int lane = threadIdx.x & 31;
  const int n    = lane & 15;              // column within 16-wide tile
  const int kg   = lane >> 4;              // lane half (K-group / M-group)
  const int d    = wave * 16 + n;          // this lane's d index
  const int mB   = kg * 8;                 // C-fragment rows: mB..mB+7

  const int    SBH = T_NH * T_NG * T_D;              // 3072 : b-stride inside Wx[t]
  const int    SMO = T_NH * T_D;                     // 768  : b-stride in out/states
  const size_t SBD = (size_t)T_B * T_NH * T_D;       // 12288: per-t plane in out
  const float* WxH = Wx + head * (T_NG * T_D);       // head base inside a timestep

  // ---- Preload recurrent weights as bf16 WMMA B-fragments (kept in VGPRs) ----
  // B (32x16 KxN), contiguous k order as in A: lanes 0-15 hold K=0..15, lanes 16-31 hold
  // K=16..31; VGPR v packs K=2v,2v+1; column N = lane&15.
  v16bf Bf[4][2];
#pragma unroll
  for (int g = 0; g < 4; ++g) {
    const float* Rg = R + (size_t)(head * T_NG + g) * T_D * T_D;  // [k][e] row-major
#pragma unroll
    for (int kc = 0; kc < 2; ++kc) {
#pragma unroll
      for (int v = 0; v < 8; ++v) {
        int k = kc * 32 + kg * 16 + 2 * v;
        Bf[g][kc][2 * v]     = (bf16_t)Rg[(size_t)k * T_D + d];
        Bf[g][kc][2 * v + 1] = (bf16_t)Rg[(size_t)(k + 1) * T_D + d];
      }
    }
  }
  const float bi = bias[(head * T_NG + 0) * T_D + d];
  const float bf = bias[(head * T_NG + 1) * T_D + d];
  const float bz = bias[(head * T_NG + 2) * T_D + d];
  const float bo = bias[(head * T_NG + 3) * T_D + d];

  // ---- Initial state: c in registers, h0 -> LDS buffer 0, emit t=0 outputs ----
  float creg[8];
#pragma unroll
  for (int r = 0; r < 8; ++r) {
    int m = mB + r;
    size_t si = (size_t)m * SMO + (size_t)head * T_D + d;
    float h0 = states[si];
    float c0 = states[(size_t)T_B * SMO + si];
    creg[r] = c0;
    hbuf[m * HSTR + d] = (bf16_t)h0;
    out[si] = h0;                                   // [0][t=0]
    out[(size_t)(T + 1) * SBD + si] = c0;           // [1][t=0]
  }

  // ---- Prime the Wx register pipeline with step 0 (non-temporal stream) ----
  float wxc[4][8];
#pragma unroll
  for (int g = 0; g < 4; ++g)
#pragma unroll
    for (int r = 0; r < 8; ++r)
      wxc[g][r] = __builtin_nontemporal_load(
          WxH + (size_t)(mB + r) * SBH + g * T_D + d);

  __syncthreads();

  // ---- Sequential scan over T ----
  for (int t = 0; t < T; ++t) {
    // Issue next step's Wx loads FIRST: independent of the recurrence, they
    // retire while this step's WMMA chain + gate tail execute.
    int tn = (t + 1 < T) ? (t + 1) : t;             // clamp: uniform, branchless
    const float* wxtn = WxH + (size_t)tn * T_B * SBH;
    float wxn[4][8];
#pragma unroll
    for (int g = 0; g < 4; ++g)
#pragma unroll
      for (int r = 0; r < 8; ++r)
        wxn[g][r] = __builtin_nontemporal_load(
            wxtn + (size_t)(mB + r) * SBH + g * T_D + d);

    // Prefetch two steps ahead into L2 (register pipeline covers t+1).
    if (t + 2 < T) {
      int pid = threadIdx.x;
      const float* pf = WxH + (size_t)(t + 2) * T_B * SBH
                        + (size_t)(pid >> 3) * SBH + ((pid >> 1) & 3) * T_D + (pid & 1) * 32;
      __builtin_prefetch(pf, 0, 1);
    }

    // A-fragments (16x32 bf16) from LDS.
    // Lanes 0-15 (M=lane): VGPR0-3 = K kc*32+0..7,  VGPR4-7 = K kc*32+8..15
    // Lanes 16-31(M=lane-16): VGPR0-3 = K kc*32+16..23, VGPR4-7 = K kc*32+24..31
    const bf16_t* hb = hbuf + (t & 1) * (16 * HSTR);
    v16bf A[2];
#pragma unroll
    for (int kc = 0; kc < 2; ++kc) {
      int kb = kc * 32 + kg * 16;
      v8bf lo = *(const v8bf*)(hb + n * HSTR + kb);
      v8bf hi = *(const v8bf*)(hb + n * HSTR + kb + 8);
      A[kc] = __builtin_shufflevector(lo, hi, 0, 1, 2, 3, 4, 5, 6, 7,
                                      8, 9, 10, 11, 12, 13, 14, 15);
    }

    // 8 WMMA per wave per step: 4 gates x (K=64 as 2 x 16x16x32).
    v8f acc[4];
#pragma unroll
    for (int g = 0; g < 4; ++g) {
      v8f c = {0.f, 0.f, 0.f, 0.f, 0.f, 0.f, 0.f, 0.f};
      c = __builtin_amdgcn_wmma_f32_16x16x32_bf16(false, A[0], false, Bf[g][0],
                                                  (short)0, c, false, false);
      c = __builtin_amdgcn_wmma_f32_16x16x32_bf16(false, A[1], false, Bf[g][1],
                                                  (short)0, c, false, false);
      acc[g] = c;
    }

    // Register-local LSTM cell update; every lane has all 4 gates at its (b,d).
    bf16_t* hnext = hbuf + ((t + 1) & 1) * (16 * HSTR);
    float* outh = out + (size_t)(t + 1) * SBD;
    float* outc = out + (size_t)(T + 1) * SBD + (size_t)(t + 1) * SBD;
#pragma unroll
    for (int r = 0; r < 8; ++r) {
      int m = mB + r;
      float gi = fsig  (acc[0][r] + wxc[0][r] + bi);
      float gf = fsig  (acc[1][r] + wxc[1][r] + bf);
      float gz = ftanh_(acc[2][r] + wxc[2][r] + bz);
      float go = fsig  (acc[3][r] + wxc[3][r] + bo);
      float cv = gf * creg[r] + gi * gz;
      creg[r] = cv;
      float hv = go * ftanh_(cv);
      hnext[m * HSTR + d] = (bf16_t)hv;
      size_t oi = (size_t)m * SMO + (size_t)head * T_D + d;
      __builtin_nontemporal_store(hv, outh + oi);
      __builtin_nontemporal_store(cv, outc + oi);
    }

    // Rotate the Wx pipeline (pure register renaming after unroll).
#pragma unroll
    for (int g = 0; g < 4; ++g)
#pragma unroll
      for (int r = 0; r < 8; ++r)
        wxc[g][r] = wxn[g][r];

    __syncthreads();
  }
}

extern "C" void kernel_launch(void* const* d_in, const int* in_sizes, int n_in,
                              void* d_out, int out_size, void* d_ws, size_t ws_size,
                              hipStream_t stream) {
  const float* Wx     = (const float*)d_in[0];
  const float* states = (const float*)d_in[1];
  const float* R      = (const float*)d_in[2];
  const float* b      = (const float*)d_in[3];
  float* out = (float*)d_out;
  int T = in_sizes[0] / (T_B * T_NH * T_NG * T_D);
  flashrnn_lstm_wmma<<<T_NH, 128, 0, stream>>>(Wx, states, R, b, out, T);
}
